// Dynamics_10874857193488
// MI455X (gfx1250) — matrix operand order repaired and checked by reading, not compiled, not run on hardware
//
#include <hip/hip_runtime.h>

typedef __attribute__((ext_vector_type(16))) _Float16 v16h;
typedef __attribute__((ext_vector_type(8)))  _Float16 v8h;
typedef __attribute__((ext_vector_type(8)))  float    v8f;
typedef __attribute__((ext_vector_type(4)))  int      v4i;

constexpr int B  = 32768;   // batch
constexpr int P  = 4;       // policies
constexpr int D  = 512;     // latent dim (output)
constexpr int H  = 1024;    // hidden
constexpr int IN = 516;     // latent+action
constexpr int KP = 544;     // IN padded to multiple of 32 (17*32)

// CDNA5 async global->LDS copy path (ASYNCcnt-tracked), with graceful fallback.
#if defined(__HIP_DEVICE_COMPILE__) && __has_builtin(__builtin_amdgcn_global_load_async_to_lds_b128)
#define HAVE_ASYNC 1
typedef __attribute__((address_space(1))) v4i* as1_v4i;
typedef __attribute__((address_space(3))) v4i* as3_v4i;
#else
#define HAVE_ASYNC 0
#endif

__device__ __forceinline__ void wait_async_cnt0() {
#if HAVE_ASYNC
#if __has_builtin(__builtin_amdgcn_s_wait_asynccnt)
    __builtin_amdgcn_s_wait_asynccnt(0);
#else
    asm volatile("s_wait_asynccnt 0" ::: "memory");
#endif
#endif
}

// ---------------- routing (counting sort) ----------------
// meta[0..3]=counts, meta[4..7]=segment offsets, meta[8..11]=scatter cursors
__global__ void zero_meta_k(int* meta) {
    if (threadIdx.x < 12) meta[threadIdx.x] = 0;
}

__global__ void hist_k(const int* __restrict__ idx, int* __restrict__ meta) {
    int r = blockIdx.x * 256 + threadIdx.x;
    if (r < B) atomicAdd(&meta[idx[r]], 1);
}

__global__ void offsets_k(int* meta) {
    if (threadIdx.x == 0) {
        int o = 0;
        for (int p = 0; p < P; ++p) {
            meta[4 + p] = o;
            meta[8 + p] = o;
            o += meta[p];
        }
    }
}

__global__ void scatter_k(const int* __restrict__ idx, int* __restrict__ meta,
                          int* __restrict__ row_index) {
    int r = blockIdx.x * 256 + threadIdx.x;
    if (r < B) {
        int p = idx[r];
        int pos = atomicAdd(&meta[8 + p], 1);
        row_index[pos] = r;
    }
}

// ---------------- gather + convert activations ----------------
__global__ void gather_x_k(const float* __restrict__ latents,
                           const float* __restrict__ actions,
                           const int* __restrict__ row_index,
                           _Float16* __restrict__ xg) {
    int i = blockIdx.x * 256 + threadIdx.x;     // grid sized exactly B*KP/256
    int s = i / KP;
    int k = i - s * KP;
    int r = row_index[s];
    float v = 0.0f;
    if (k < D)          v = latents[(size_t)r * D + k];
    else if (k < IN)    v = actions[(size_t)r * 4 + (k - D)];
    xg[i] = (_Float16)v;
}

// ---------------- weight conversion ----------------
__global__ void cvt_w1_k(const float* __restrict__ W1, _Float16* __restrict__ w1h) {
    int i = blockIdx.x * 256 + threadIdx.x;     // total P*KP*H
    if (i >= P * KP * H) return;
    int n = i & (H - 1);
    int t = i >> 10;            // p*KP + k
    int k = t % KP;
    int p = t / KP;
    _Float16 v = (_Float16)0.0f;
    if (k < IN) v = (_Float16)W1[((size_t)p * IN + k) * H + n];
    w1h[i] = v;
}

__global__ void cvt_f16_k(const float* __restrict__ src, _Float16* __restrict__ dst, int n) {
    int i = blockIdx.x * 256 + threadIdx.x;
    if (i < n) dst[i] = (_Float16)src[i];
}

// ---------------- per-policy GEMM with WMMA ----------------
// Tile: 128(M) x 128(N) x 32(K), double-buffered LDS, async A staging.
// 8 waves: 2(M) x 4(N); each wave: 64x32 sub-tile = 4x2 WMMA f32_16x16x32_f16 accs.
template <int K, int N, bool RELU, bool SCATTER>
__global__ __launch_bounds__(256) void gemm_moe_k(
    const _Float16* __restrict__ A,       // [B, K] rows in sorted order
    const _Float16* __restrict__ W,       // [P, K, N] f16
    const float*    __restrict__ bias,    // [P, N]
    _Float16*       __restrict__ Hout,    // [B, N] sorted order (if !SCATTER)
    float*          __restrict__ Oout,    // [B, N] original order (if SCATTER)
    const int*      __restrict__ meta,    // counts[4], offs[4]
    const int*      __restrict__ row_index) {
    const int p     = blockIdx.z;
    const int cnt   = meta[p];
    const int mtile = blockIdx.y;
    if (mtile * 128 >= cnt) return;       // empty tile for this policy
    const int segBase = meta[4 + p];
    const int ntile   = blockIdx.x;

    // Double-buffered tiles. Bt is the weight tile stored N-major (transposed)
    // so a lane's 16 B-matrix K-elements are contiguous (2 x ds_load_b128).
    __shared__ __align__(16) _Float16 As[2][128][40];
    __shared__ __align__(16) _Float16 Bt[2][128][40];

    const int tid  = threadIdx.x;
    const int lane = tid & 31;
    const int wave = tid >> 5;
    const int wm = (wave >> 2) * 64;      // wave M offset: 0 / 64
    const int wn = (wave & 3) * 32;       // wave N offset: 0 / 32 / 64 / 96
    const int g  = lane >> 4;             // half-wave group
    const int lr = lane & 15;

    const _Float16* Wp = W + (size_t)p * K * N;

    // Zero A rows past count[p] once; they are never re-staged, so they stay 0.
#pragma unroll
    for (int it = 0; it < 2; ++it) {
        int c   = tid + it * 256;
        int row = c >> 2;
        int kc  = (c & 3) << 3;
        if (mtile * 128 + row >= cnt) {
            uint4 z; z.x = z.y = z.z = z.w = 0u;
            *(uint4*)&As[0][row][kc] = z;
            *(uint4*)&As[1][row][kc] = z;
        }
    }

    // Stage activation tile (global -> LDS, async when available).
    auto stageA = [&](int kt, int buf) {
#pragma unroll
        for (int it = 0; it < 2; ++it) {
            int c   = tid + it * 256;
            int row = c >> 2;
            int kc  = (c & 3) << 3;
            int m   = mtile * 128 + row;
            if (m < cnt) {
                const _Float16* gp = A + (size_t)(segBase + m) * K + (kt + kc);
#if HAVE_ASYNC
                __builtin_amdgcn_global_load_async_to_lds_b128(
                    (as1_v4i)(unsigned long long)gp,
                    (as3_v4i)&As[buf][row][kc],
                    0, 0);
#else
                *(uint4*)&As[buf][row][kc] = *(const uint4*)gp;
#endif
            }
        }
    };

    // Stage weight tile transposed: Bt[n][k] = W[kt+k][ntile*128+n].
    auto stageB = [&](int kt, int buf) {
#pragma unroll
        for (int it = 0; it < 2; ++it) {
            int c  = tid + it * 256;
            int kk = c >> 4;              // 0..31
            int n0 = (c & 15) << 3;       // 0..120
            v8h v = *(const v8h*)(Wp + (size_t)(kt + kk) * N + (ntile * 128 + n0));
#pragma unroll
            for (int j = 0; j < 8; ++j) Bt[buf][n0 + j][kk] = v[j];
        }
    };

    v8f acc[4][2] = {};

    stageA(0, 0);
    stageB(0, 0);
    wait_async_cnt0();
    __syncthreads();

    int buf = 0;
    for (int kt = 0; kt < K; kt += 32) {
        if (kt + 32 < K) {                // prefetch next tile into other buffer
            stageA(kt + 32, buf ^ 1);
            stageB(kt + 32, buf ^ 1);
        }

        // A fragments (ISA 16-bit A 16x32: group g holds K=g*8+0..7, 16+g*8+0..7)
        v16h af[4];
#pragma unroll
        for (int mi = 0; mi < 4; ++mi) {
            const int r = wm + mi * 16 + lr;
            const v8h a0 = *(const v8h*)&As[buf][r][g * 8];
            const v8h a1 = *(const v8h*)&As[buf][r][g * 8 + 16];
#pragma unroll
            for (int j = 0; j < 8; ++j) { af[mi][j] = a0[j]; af[mi][8 + j] = a1[j]; }
        }
        // B fragments (group g holds K=g*8+0..7, 16+g*8+0..7; N = lane&15)
        v16h bf[2];
#pragma unroll
        for (int ni = 0; ni < 2; ++ni) {
            const int cidx = wn + ni * 16 + lr;
            const v8h b0 = *(const v8h*)&Bt[buf][cidx][g * 8];
            const v8h b1 = *(const v8h*)&Bt[buf][cidx][g * 8 + 16];
#pragma unroll
            for (int j = 0; j < 8; ++j) { bf[ni][j] = b0[j]; bf[ni][8 + j] = b1[j]; }
        }
#pragma unroll
        for (int mi = 0; mi < 4; ++mi)
#pragma unroll
            for (int ni = 0; ni < 2; ++ni)
                acc[mi][ni] = __builtin_amdgcn_wmma_f32_16x16x32_f16(
                    false, af[mi], false, bf[ni], (short)0, acc[mi][ni], false, false);

        wait_async_cnt0();                // next-buffer async copies complete
        __syncthreads();                  // + all waves' Bt stores visible
        buf ^= 1;
    }

    // Epilogue: bias (+ReLU), store f16 hidden or scatter f32 output.
#pragma unroll
    for (int mi = 0; mi < 4; ++mi) {
#pragma unroll
        for (int ni = 0; ni < 2; ++ni) {
            const int ncol = ntile * 128 + wn + ni * 16 + lr;
            const float bv = bias[(size_t)p * N + ncol];
#pragma unroll
            for (int i = 0; i < 8; ++i) {
                const int mrow = mtile * 128 + wm + mi * 16 + g * 8 + i;
                if (mrow < cnt) {
                    float v = acc[mi][ni][i] + bv;
                    if (RELU) v = fmaxf(v, 0.0f);
                    if (SCATTER) {
                        const int r = row_index[segBase + mrow];
                        Oout[(size_t)r * N + ncol] = v;
                    } else {
                        Hout[(size_t)(segBase + mrow) * N + ncol] = (_Float16)v;
                    }
                }
            }
        }
    }
}

// ---------------- launcher ----------------
extern "C" void kernel_launch(void* const* d_in, const int* in_sizes, int n_in,
                              void* d_out, int out_size, void* d_ws, size_t ws_size,
                              hipStream_t stream) {
    const float* latents = (const float*)d_in[0];
    const int*   pidx    = (const int*)  d_in[1];
    const float* actions = (const float*)d_in[2];
    const float* W1      = (const float*)d_in[3];
    const float* b1      = (const float*)d_in[4];
    const float* W2      = (const float*)d_in[5];
    const float* b2      = (const float*)d_in[6];
    const float* W3      = (const float*)d_in[7];
    const float* b3      = (const float*)d_in[8];
    float*       out     = (float*)d_out;

    char* ws = (char*)d_ws;
    int* meta      = (int*)ws;                            // 12 ints
    int* row_index = (int*)(ws + 256);                    // B ints
    size_t off = 256 + (size_t)B * 4;
    off = (off + 255) & ~(size_t)255;
    _Float16* xg  = (_Float16*)(ws + off); off += (size_t)B * KP * 2;
    _Float16* h1  = (_Float16*)(ws + off); off += (size_t)B * H  * 2;
    _Float16* h2  = (_Float16*)(ws + off); off += (size_t)B * H  * 2;
    _Float16* w1h = (_Float16*)(ws + off); off += (size_t)P * KP * H * 2;
    _Float16* w2h = (_Float16*)(ws + off); off += (size_t)P * H  * H * 2;
    _Float16* w3h = (_Float16*)(ws + off); off += (size_t)P * H  * D * 2;
    (void)ws_size; (void)in_sizes; (void)n_in; (void)out_size;

    // 1) routing: counting sort of rows by policy
    zero_meta_k<<<1, 32, 0, stream>>>(meta);
    hist_k<<<B / 256, 256, 0, stream>>>(pidx, meta);
    offsets_k<<<1, 1, 0, stream>>>(meta);
    scatter_k<<<B / 256, 256, 0, stream>>>(pidx, meta, row_index);

    // 2) gather inputs (K padded 516->544) and convert weights to f16
    gather_x_k<<<(B * KP) / 256, 256, 0, stream>>>(latents, actions, row_index, xg);
    cvt_w1_k<<<(P * KP * H + 255) / 256, 256, 0, stream>>>(W1, w1h);
    cvt_f16_k<<<(P * H * H) / 256, 256, 0, stream>>>(W2, w2h, P * H * H);
    cvt_f16_k<<<(P * H * D) / 256, 256, 0, stream>>>(W3, w3h, P * H * D);

    // 3) three per-policy GEMMs (grid.y covers worst case; blocks early-exit)
    dim3 blk(256);
    gemm_moe_k<KP, H, true,  false><<<dim3(H / 128, B / 128, P), blk, 0, stream>>>(
        xg, w1h, b1, h1, nullptr, meta, row_index);
    gemm_moe_k<H,  H, true,  false><<<dim3(H / 128, B / 128, P), blk, 0, stream>>>(
        h1, w2h, b2, h2, nullptr, meta, row_index);
    gemm_moe_k<H,  D, false, true ><<<dim3(D / 128, B / 128, P), blk, 0, stream>>>(
        h2, w3h, b3, nullptr, out, meta, row_index);
}
